// BoxMultiHeadedAttention_42425686950461
// MI455X (gfx1250) — compile-verified, matrix operand order checked by reading, not run on hardware
//
#include <hip/hip_runtime.h>
#include <hip/hip_bf16.h>

// ---------------------------------------------------------------------------
// BoxMultiHeadedAttention for MI455X (gfx1250, wave32, WMMA bf16 16x16x32).
// B=32, N=196, D=1024, H=8, dk=128.  ~58 GFLOP of GEMM vs ~120MB traffic
// => compute bound => all four GEMMs run on v_wmma_f32_16x16x32_bf16.
// All fragment loads are unconditional b128; boundary masking is arithmetic
// (x * {0,1}) or value-select AFTER unconditional loads, so no exec-mask
// branch serialization anywhere in the hot paths.
// ---------------------------------------------------------------------------

#define B_  32
#define N_  196
#define D_  1024
#define H_  8
#define DK  128
#define NP  224   // padded sequence rows per (b,h)  (7 * 32)
#define MS  208   // score row length (13 * 16)

typedef __attribute__((ext_vector_type(16))) __bf16 v16bf;
typedef __attribute__((ext_vector_type(8)))  float  v8f;

struct B32x8 { uint4 lo, hi; };   // 32 bytes == v16bf

static __device__ __forceinline__ v8f wmma_bf16(v16bf a, v16bf b, v8f c) {
  // (neg_a, A, neg_b, B, c_mod, C, reuse_a, reuse_b)
  return __builtin_amdgcn_wmma_f32_16x16x32_bf16(false, a, false, b, (short)0, c,
                                                 false, false);
}

// 16-bit 16x32 fragment layout: lane holds row (lane&15); register v holds
//   k = k0 + 8*lane[4] + (v<4 ? 2v : 16+2(v-4)) .. +1
// => regs 0..3 are 8 consecutive bf16 (16B), regs 4..7 the next 8 at +16.

// bf16 source: exactly two b128 loads, bit-cast straight into the fragment.
static __device__ __forceinline__ v16bf load_frag_bf16(const __bf16* rowp,
                                                       int k0, int lane) {
  int base = k0 + ((lane >> 4) << 3);
  B32x8 r;
  r.lo = *(const uint4*)(rowp + base);
  r.hi = *(const uint4*)(rowp + base + 16);
  return __builtin_bit_cast(v16bf, r);
}

// f32 source: four b128 loads + pk conversions; msk is 0.0f or 1.0f
// (1.0f folds away; 0.0f zeroes pad rows without any branch).
static __device__ __forceinline__ v16bf load_frag_f32(const float* rowp,
                                                      int k0, int lane, float msk) {
  int base = k0 + ((lane >> 4) << 3);
  float4 a0 = *(const float4*)(rowp + base);
  float4 a1 = *(const float4*)(rowp + base + 4);
  float4 a2 = *(const float4*)(rowp + base + 16);
  float4 a3 = *(const float4*)(rowp + base + 20);
  v16bf f;
  f[0]  = (__bf16)(a0.x * msk); f[1]  = (__bf16)(a0.y * msk);
  f[2]  = (__bf16)(a0.z * msk); f[3]  = (__bf16)(a0.w * msk);
  f[4]  = (__bf16)(a1.x * msk); f[5]  = (__bf16)(a1.y * msk);
  f[6]  = (__bf16)(a1.z * msk); f[7]  = (__bf16)(a1.w * msk);
  f[8]  = (__bf16)(a2.x * msk); f[9]  = (__bf16)(a2.y * msk);
  f[10] = (__bf16)(a2.z * msk); f[11] = (__bf16)(a2.w * msk);
  f[12] = (__bf16)(a3.x * msk); f[13] = (__bf16)(a3.y * msk);
  f[14] = (__bf16)(a3.z * msk); f[15] = (__bf16)(a3.w * msk);
  return f;
}

// ---------------------------------------------------------------------------
// Kernel 1: box geometry embedding -> log(clip(relu(geo @ WG^T + b), 1e-6)).
// One block per (b, n); threads cover m (0..207). m in [196,208) -> -1e30
// so the attention softmax needs no explicit column masking.
// ---------------------------------------------------------------------------
__global__ __launch_bounds__(256) void geo_logw_kernel(
    const float* __restrict__ box, const float* __restrict__ WGw,
    const float* __restrict__ WGb, float* __restrict__ logw) {
  __shared__ float Wl[H_ * 64 + H_];
  int tid = threadIdx.x;
  for (int t = tid; t < H_ * 64 + H_; t += 256)
    Wl[t] = (t < H_ * 64) ? WGw[t] : WGb[t - H_ * 64];
  __syncthreads();

  int b = blockIdx.x / N_;
  int n = blockIdx.x % N_;
  int m = tid;
  if (m >= MS) return;

  if (m >= N_) {  // padded columns: -inf-like logits
#pragma unroll
    for (int h = 0; h < H_; ++h)
      logw[(((size_t)(b * H_ + h)) * N_ + n) * MS + m] = -1.0e30f;
    return;
  }

  const float4 bn = ((const float4*)box)[b * N_ + n];
  const float4 bm = ((const float4*)box)[b * N_ + m];
  float cxn = (bn.x + bn.z) * 0.5f, cyn = (bn.y + bn.w) * 0.5f;
  float wn = bn.z - bn.x + 1.0f,    hn = bn.w - bn.y + 1.0f;
  float cxm = (bm.x + bm.z) * 0.5f, cym = (bm.y + bm.w) * 0.5f;
  float wm = bm.z - bm.x + 1.0f,    hm = bm.w - bm.y + 1.0f;

  float pos[4];
  pos[0] = __logf(fmaxf(fabsf((cxn - cxm) / wn), 1e-3f));
  pos[1] = __logf(fmaxf(fabsf((cyn - cym) / hn), 1e-3f));
  pos[2] = __logf(wn / wm);
  pos[3] = __logf(hn / hm);

  // 1000^(-f/8), f = 0..7
  const float dimf[8] = {1.0f,          0.4216965034f, 0.1778279410f, 0.0749894209f,
                         0.0316227766f, 0.0133352143f, 0.0056234133f, 0.0023713737f};

  float acc[H_];
#pragma unroll
  for (int h = 0; h < H_; ++h) acc[h] = Wl[H_ * 64 + h];

#pragma unroll
  for (int p = 0; p < 4; ++p) {
    float base100 = 100.0f * pos[p];
#pragma unroll
    for (int f = 0; f < 8; ++f) {
      float ang = base100 * dimf[f];
      float s = __sinf(ang);
      float c = __cosf(ang);
      int g = p * 8 + f;
#pragma unroll
      for (int h = 0; h < H_; ++h)  // LDS broadcast reads: conflict-free
        acc[h] += s * Wl[h * 64 + g] + c * Wl[h * 64 + 32 + g];
    }
  }

#pragma unroll
  for (int h = 0; h < H_; ++h) {
    float wg = fmaxf(acc[h], 0.0f);
    logw[(((size_t)(b * H_ + h)) * N_ + n) * MS + m] = __logf(fmaxf(wg, 1e-6f));
  }
}

// ---------------------------------------------------------------------------
// Kernel 2: fused Q/K/V projection, X @ W^T + b, bf16 WMMA with f32 acc.
// grid = (colgrp 8, b*7 rowgrps, mode 3). Block = 256 thr = 8 waves,
// each wave owns two 16x16 C tiles of a 32x128 block tile; K loop = 32 x 32.
// Q/K stored (b,h,n_pad,dk) row-major bf16; V stored transposed (b,h,dk,n_pad)
// so the attn@V B-fragment is a contiguous-K row load. Pad rows = 0.
// ---------------------------------------------------------------------------
__global__ __launch_bounds__(256) void proj_qkv_kernel(
    const float* __restrict__ Xq, const float* __restrict__ Xk,
    const float* __restrict__ Xv,
    const float* __restrict__ Wq, const float* __restrict__ bq,
    const float* __restrict__ Wk, const float* __restrict__ bk,
    const float* __restrict__ Wv, const float* __restrict__ bv,
    __bf16* __restrict__ Qh, __bf16* __restrict__ Kh, __bf16* __restrict__ Vt) {
  int mode = blockIdx.z;              // 0=Q 1=K 2=V
  int b    = blockIdx.y / 7;
  int nt   = blockIdx.y % 7;
  int c0   = blockIdx.x * 128;
  int tid = threadIdx.x, w = tid >> 5, lane = tid & 31;
  int i = w >> 2, jc = w & 3;

  const float* X    = (mode == 0) ? Xq : (mode == 1) ? Xk : Xv;
  const float* W    = (mode == 0) ? Wq : (mode == 1) ? Wk : Wv;
  const float* bias = (mode == 0) ? bq : (mode == 1) ? bk : bv;

  int nl = nt * 32 + 16 * i + (lane & 15);  // padded local row (0..223)
  float amsk = (nl < N_) ? 1.0f : 0.0f;     // arithmetic row mask (no branches)
  const float* Arow  = X + ((size_t)b * N_ + (nl < N_ ? nl : 0)) * D_;
  const float* Brow0 = W + (size_t)(c0 + 16 * jc + (lane & 15)) * D_;
  const float* Brow1 = Brow0 + (size_t)64 * D_;

  v8f acc0, acc1;
#pragma unroll
  for (int e = 0; e < 8; ++e) { acc0[e] = 0.0f; acc1[e] = 0.0f; }

  for (int kc = 0; kc < D_ / 32; ++kc) {
    int k0 = kc * 32;
    v16bf af  = load_frag_f32(Arow, k0, lane, amsk);
    v16bf bf0 = load_frag_f32(Brow0, k0, lane, 1.0f);
    v16bf bf1 = load_frag_f32(Brow1, k0, lane, 1.0f);
    acc0 = wmma_bf16(af, bf0, acc0);
    acc1 = wmma_bf16(af, bf1, acc1);
  }

  // epilogue: bias loads hoisted & unconditional (c is always in range),
  // row-validity applied as a value select only.
  int cA = c0 + 16 * jc + (lane & 15);
  int cB = cA + 64;
  float bsA = bias[cA];
  float bsB = bias[cB];
  int hA = cA >> 7, ddA = cA & 127;
  int hB = cB >> 7, ddB = cB & 127;

  int rb = nt * 32 + 16 * i + 8 * (lane >> 4);
#pragma unroll
  for (int r = 0; r < 8; ++r) {
    int n = rb + r;                 // padded local row
    bool ok = n < N_;
    float vA = ok ? (acc0[r] + bsA) : 0.0f;
    float vB = ok ? (acc1[r] + bsB) : 0.0f;
    if (mode == 2) {
      Vt[(((size_t)b * H_ + hA) * DK + ddA) * NP + n] = (__bf16)vA;
      Vt[(((size_t)b * H_ + hB) * DK + ddB) * NP + n] = (__bf16)vB;
    } else if (mode == 1) {
      Kh[(((size_t)b * H_ + hA) * NP + n) * DK + ddA] = (__bf16)vA;
      Kh[(((size_t)b * H_ + hB) * NP + n) * DK + ddB] = (__bf16)vB;
    } else {
      Qh[(((size_t)b * H_ + hA) * NP + n) * DK + ddA] = (__bf16)vA;
      Qh[(((size_t)b * H_ + hB) * NP + n) * DK + ddB] = (__bf16)vB;
    }
  }
}

// ---------------------------------------------------------------------------
// Kernel 3: attention core. One block per (b, h, 32-row slab of n).
// scores = Q K^T (WMMA, Q frags register-cached) * 1/sqrt(dk) + logw -> LDS,
// shuffle softmax (8 lanes/row), bf16 P in LDS, O = P V (WMMA, A from LDS).
// ---------------------------------------------------------------------------
__global__ __launch_bounds__(256) void attn_kernel(
    const __bf16* __restrict__ Qh, const __bf16* __restrict__ Kh,
    const __bf16* __restrict__ Vt, const float* __restrict__ logw,
    __bf16* __restrict__ Oc) {
  __shared__ float  S[32][MS];                    // 26.6 KB
  __shared__ __align__(16) __bf16 P[32][NP];      // 14.3 KB

  int blk = blockIdx.x;
  int b   = blk / (H_ * 7);
  int rem = blk % (H_ * 7);
  int h   = rem / 7;
  int n0  = (rem % 7) * 32;
  int tid = threadIdx.x, w = tid >> 5, lane = tid & 31;
  int i = w >> 2, jc = w & 3;

  const __bf16* Qb = Qh + (((size_t)b * H_ + h) * NP) * DK;
  const __bf16* Kb = Kh + (((size_t)b * H_ + h) * NP) * DK;
  const __bf16* Vb = Vt + (((size_t)b * H_ + h) * DK) * NP;
  const float*  Lb = logw + (((size_t)b * H_ + h) * N_) * MS;

  // ---- scores: S[32][208] = Q K^T / sqrt(dk) + logw -----------------------
  int rowq = n0 + 16 * i + (lane & 15);
  v16bf qf[4];
#pragma unroll
  for (int kc = 0; kc < 4; ++kc)
    qf[kc] = load_frag_bf16(Qb + (size_t)rowq * DK, kc * 32, lane);

  for (int j = jc; j < 13; j += 4) {
    v8f acc;
#pragma unroll
    for (int e = 0; e < 8; ++e) acc[e] = 0.0f;
    const __bf16* Krow = Kb + (size_t)(16 * j + (lane & 15)) * DK;
#pragma unroll
    for (int kc = 0; kc < 4; ++kc) {
      v16bf kf = load_frag_bf16(Krow, kc * 32, lane);
      acc = wmma_bf16(qf[kc], kf, acc);
    }
    int mcol = 16 * j + (lane & 15);
    int rb   = 16 * i + 8 * (lane >> 4);
#pragma unroll
    for (int r = 0; r < 8; ++r) {
      int nl = rb + r;
      int ng = n0 + nl;
      // clamped unconditional load + select (no exec-mask branching)
      int ngc = ng < N_ ? ng : (N_ - 1);
      float lw = Lb[(size_t)ngc * MS + mcol];
      lw = (ng < N_) ? lw : 0.0f;
      S[nl][mcol] = acc[r] * 0.08838834764831843f + lw;
    }
  }
  __syncthreads();

  // ---- softmax over m (8 lanes cooperate per row) -------------------------
  {
    int r = tid >> 3, j8 = tid & 7;
    float mx = -3.0e38f;
    for (int m = j8; m < MS; m += 8) mx = fmaxf(mx, S[r][m]);
    mx = fmaxf(mx, __shfl_xor(mx, 1, 8));
    mx = fmaxf(mx, __shfl_xor(mx, 2, 8));
    mx = fmaxf(mx, __shfl_xor(mx, 4, 8));
    float sum = 0.0f;
    for (int m = j8; m < MS; m += 8) {
      float e = __expf(S[r][m] - mx);
      S[r][m] = e;
      sum += e;
    }
    sum += __shfl_xor(sum, 1, 8);
    sum += __shfl_xor(sum, 2, 8);
    sum += __shfl_xor(sum, 4, 8);
    float inv = 1.0f / (sum + 1e-20f);
    for (int m = j8; m < MS; m += 8) P[r][m] = (__bf16)(S[r][m] * inv);
    for (int m = MS + j8; m < NP; m += 8) P[r][m] = (__bf16)0.0f;  // K-pad = 0
  }
  __syncthreads();

  // ---- O = P @ V  (A from LDS, B = V^T rows, K over m = 7 x 32) ----------
#pragma unroll
  for (int t = 0; t < 2; ++t) {
    int j = jc + 4 * t;  // 0..7 -> 128 output columns
    v8f acc;
#pragma unroll
    for (int e = 0; e < 8; ++e) acc[e] = 0.0f;
    const __bf16* Prow = &P[16 * i + (lane & 15)][0];
    const __bf16* Vrow = Vb + (size_t)(16 * j + (lane & 15)) * NP;
#pragma unroll
    for (int mc = 0; mc < 7; ++mc) {
      v16bf pf = load_frag_bf16(Prow, mc * 32, lane);
      v16bf vf = load_frag_bf16(Vrow, mc * 32, lane);
      acc = wmma_bf16(pf, vf, acc);
    }
    int d  = 16 * j + (lane & 15);
    int rb = 16 * i + 8 * (lane >> 4);
#pragma unroll
    for (int r = 0; r < 8; ++r) {
      int ng = n0 + rb + r;
      if (ng < N_)
        Oc[((size_t)b * N_ + ng) * D_ + h * DK + d] = (__bf16)acc[r];
    }
  }
}

// ---------------------------------------------------------------------------
// Kernel 4: output projection, out = O @ Wo^T + bo, f32 result.
// M = B*N = 6272 = 196 * 32 exactly (no padding needed).
// ---------------------------------------------------------------------------
__global__ __launch_bounds__(256) void out_proj_kernel(
    const __bf16* __restrict__ Oc, const float* __restrict__ Wo,
    const float* __restrict__ bo, float* __restrict__ out) {
  int r0 = blockIdx.y * 32;
  int c0 = blockIdx.x * 128;
  int tid = threadIdx.x, w = tid >> 5, lane = tid & 31;
  int i = w >> 2, jc = w & 3;

  const __bf16* Arow = Oc + (size_t)(r0 + 16 * i + (lane & 15)) * D_;
  const float* B0 = Wo + (size_t)(c0 + 16 * jc + (lane & 15)) * D_;
  const float* B1 = B0 + (size_t)64 * D_;

  v8f acc0, acc1;
#pragma unroll
  for (int e = 0; e < 8; ++e) { acc0[e] = 0.0f; acc1[e] = 0.0f; }

  for (int kc = 0; kc < D_ / 32; ++kc) {
    int k0 = kc * 32;
    v16bf af  = load_frag_bf16(Arow, k0, lane);
    v16bf bf0 = load_frag_f32(B0, k0, lane, 1.0f);
    v16bf bf1 = load_frag_f32(B1, k0, lane, 1.0f);
    acc0 = wmma_bf16(af, bf0, acc0);
    acc1 = wmma_bf16(af, bf1, acc1);
  }

  int cA = c0 + 16 * jc + (lane & 15);
  int cB = cA + 64;
  float bsA = bo[cA];
  float bsB = bo[cB];
  int rb = r0 + 16 * i + 8 * (lane >> 4);
#pragma unroll
  for (int r = 0; r < 8; ++r) {
    int row = rb + r;
    out[(size_t)row * D_ + cA] = acc0[r] + bsA;
    out[(size_t)row * D_ + cB] = acc1[r] + bsB;
  }
}

// ---------------------------------------------------------------------------
extern "C" void kernel_launch(void* const* d_in, const int* in_sizes, int n_in,
                              void* d_out, int out_size, void* d_ws, size_t ws_size,
                              hipStream_t stream) {
  const float* Xq  = (const float*)d_in[0];
  const float* Xk  = (const float*)d_in[1];
  const float* Xv  = (const float*)d_in[2];
  const float* box = (const float*)d_in[3];
  const float* Wq  = (const float*)d_in[4];
  const float* bq  = (const float*)d_in[5];
  const float* Wk  = (const float*)d_in[6];
  const float* bk  = (const float*)d_in[7];
  const float* Wv  = (const float*)d_in[8];
  const float* bv  = (const float*)d_in[9];
  const float* Wo  = (const float*)d_in[10];
  const float* bo  = (const float*)d_in[11];
  const float* WGw = (const float*)d_in[12];
  const float* WGb = (const float*)d_in[13];

  // workspace carve-up (~98.6 MB total)
  char* ws = (char*)d_ws;
  size_t szQK = (size_t)B_ * H_ * NP * DK * sizeof(__bf16);   // 14,680,064 B
  size_t szL  = (size_t)B_ * H_ * N_ * MS * sizeof(float);    // 41,746,432 B
  __bf16* Qh   = (__bf16*)(ws);
  __bf16* Kh   = (__bf16*)(ws + szQK);
  __bf16* Vt   = (__bf16*)(ws + 2 * szQK);
  float*  logw = (float*)(ws + 3 * szQK);
  __bf16* Oc   = (__bf16*)(ws + 3 * szQK + szL);

  geo_logw_kernel<<<dim3(B_ * N_), 256, 0, stream>>>(box, WGw, WGb, logw);
  proj_qkv_kernel<<<dim3(8, B_ * 7, 3), 256, 0, stream>>>(
      Xq, Xk, Xv, Wq, bq, Wk, bk, Wv, bv, Qh, Kh, Vt);
  attn_kernel<<<dim3(B_ * H_ * 7), 256, 0, stream>>>(Qh, Kh, Vt, logw, Oc);
  out_proj_kernel<<<dim3(8, (B_ * N_) / 32), 256, 0, stream>>>(Oc, Wo, bo,
                                                               (float*)d_out);
}
